// MolecularMambaBiMamba_62216896250089
// MI455X (gfx1250) — matrix operand order repaired and checked by reading, not compiled, not run on hardware
//
#include <hip/hip_runtime.h>
#include <hip/hip_bf16.h>
#include <math.h>

// ---------------------------------------------------------------------------
// Types for CDNA5 WMMA (wave32): A/B = 16 bf16 per lane, C/D = 8 f32 per lane
// ---------------------------------------------------------------------------
typedef __attribute__((ext_vector_type(16))) __bf16 v16bf;
typedef __attribute__((ext_vector_type(8)))  float  v8f;

// ---------------------------------------------------------------------------
// Fragment loaders, contiguous k order k = 16*hi + s for A and B (fp32 -> bf16 convert)
// A (16x32, MxK): lane m<16 row m, K {0..15}; lane m+16 K {16..31}
// ---------------------------------------------------------------------------
__device__ inline v16bf load_a_frag(const float* __restrict__ A, int lda,
                                    int m0, int k0, int lane) {
  const int m  = m0 + (lane & 15);
  const int hi = (lane >> 4) & 1;
  const float* row = A + (size_t)m * lda + k0 + hi * 16;
  v16bf a;
#pragma unroll
  for (int v = 0; v < 8; ++v) {
    const int kb = 2 * v;
    float2 p = *(const float2*)(row + kb);
    a[2 * v]     = (__bf16)p.x;
    a[2 * v + 1] = (__bf16)p.y;
  }
  return a;
}

// B (32x16, KxN), built from W (N x K row-major, since out = x @ W^T):
// lane n<16 col n, VGPR v holds K=2v,2v+1; lanes 16-31 K += 16
__device__ inline v16bf load_b_frag(const float* __restrict__ W, int ldw,
                                    int n0, int k0, int N, int lane) {
  const int n  = n0 + (lane & 15);
  const int hi = (lane >> 4) & 1;
  v16bf b;
  if (n < N) {
    const float* row = W + (size_t)n * ldw + k0 + hi * 16;
#pragma unroll
    for (int v = 0; v < 8; ++v) {
      float2 p = *(const float2*)(row + 2 * v);
      b[2 * v]     = (__bf16)p.x;
      b[2 * v + 1] = (__bf16)p.y;
    }
  } else {
#pragma unroll
    for (int v = 0; v < 16; ++v) b[v] = (__bf16)0.f;
  }
  return b;
}

// ---------------------------------------------------------------------------
// Generic GEMM:  out = epilogue( A[MxK] @ W[NxK]^T )
//   bias  : optional per-col bias
//   resid/scale : out = resid + acc*scale[col]   (read/write same element)
//   ldo/col_off : output row stride & column offset (for concat buffers)
//   flipL : if >0, output row (b*flipL+l) -> (b*flipL + flipL-1-l)
// Each wave computes a 16(M) x 64(N) strip: 4 WMMAs per 32-wide k-step,
// reusing the A fragment.  K must be a multiple of 32 (true for all calls).
// ---------------------------------------------------------------------------
__global__ __launch_bounds__(128)
void gemm_bf16_wmma(const float* __restrict__ A, const float* __restrict__ W,
                    const float* __restrict__ bias,
                    const float* __restrict__ resid,
                    const float* __restrict__ scale,
                    float* __restrict__ out,
                    int M, int N, int K, int ldo, int col_off, int flipL) {
  const int lane = threadIdx.x & 31;
  const int wave = threadIdx.x >> 5;
  const int m0 = (blockIdx.y * 4 + wave) * 16;
  const int n0 = blockIdx.x * 64;
  if (m0 >= M) return;   // wave-uniform: EXEC stays all-1s for WMMA

  v8f acc[4];
#pragma unroll
  for (int t = 0; t < 4; ++t)
#pragma unroll
    for (int r = 0; r < 8; ++r) acc[t][r] = 0.f;

  for (int k0 = 0; k0 < K; k0 += 32) {
    if (k0 + 32 < K)
      __builtin_prefetch(A + (size_t)(m0 + (lane & 15)) * K + k0 + 32, 0, 1);
    v16bf a = load_a_frag(A, K, m0, k0, lane);
#pragma unroll
    for (int t = 0; t < 4; ++t) {
      v16bf b = load_b_frag(W, K, n0 + 16 * t, k0, N, lane);
      acc[t] = __builtin_amdgcn_wmma_f32_16x16x32_bf16(
          false, a, false, b, (short)0, acc[t], false, false);
    }
  }

  // C/D layout: element r -> row m0 + r + 8*hi, col = n0 + 16*t + (lane&15)
  const int hi   = (lane >> 4) & 1;
  const int colb = lane & 15;
#pragma unroll
  for (int t = 0; t < 4; ++t) {
    const int n = n0 + 16 * t + colb;
    if (n >= N) continue;
    const float bv = bias ? bias[n] : 0.f;
#pragma unroll
    for (int r = 0; r < 8; ++r) {
      const int m = m0 + r + 8 * hi;
      float v = acc[t][r] + bv;
      int orow = m;
      if (flipL > 0) {
        const int bb = m / flipL, ll = m % flipL;
        orow = bb * flipL + (flipL - 1 - ll);
      }
      const size_t idx = (size_t)orow * ldo + col_off + n;
      if (resid) v = resid[idx] + v * scale[n];
      out[idx] = v;
    }
  }
}

// ---------------------------------------------------------------------------
// Fused Mamba2 inner kernel: causal conv(4) + SiLU + softplus(dt) + SSM scan
// + D skip + gated RMSNorm.  One block per sequence; thread = one channel of
// d_inner (h = tid>>6, p = tid&63); 16 f32 of SSM state in registers.
// zx row layout: [ z(DI) | xBC(DI+32) | dt(NH) ], LDZ = 2*DI + 32 + NH.
// ---------------------------------------------------------------------------
template <int DI, int NSEQ, int LDZ, int NH>
__global__ __launch_bounds__(DI)
void scan_kernel(const float* __restrict__ zx, const float* __restrict__ conv_w,
                 const float* __restrict__ conv_b,
                 const float* __restrict__ dt_bias,
                 const float* __restrict__ A_log, const float* __restrict__ Dp,
                 const float* __restrict__ norm_w, float* __restrict__ yn) {
  constexpr int NS = 16;
  constexpr int EXTRA = 2 * NS;            // B + C channels
  const int seq = blockIdx.x;
  const int tid = threadIdx.x;
  const int h = tid >> 6;

  __shared__ float xsh[DI + EXTRA];
  __shared__ float dts[NH];
  __shared__ float dAs[NH];
  __shared__ float red[DI / 32];

  float st[NS];
#pragma unroll
  for (int n = 0; n < NS; ++n) st[n] = 0.f;

  const float Dh = Dp[h];
  const float nw = norm_w[tid];
  float w0[4], b0 = conv_b[tid];
#pragma unroll
  for (int i = 0; i < 4; ++i) w0[i] = conv_w[tid * 4 + i];
  float w1[4] = {0.f, 0.f, 0.f, 0.f}, b1 = 0.f;
  if (tid < EXTRA) {
#pragma unroll
    for (int i = 0; i < 4; ++i) w1[i] = conv_w[(DI + tid) * 4 + i];
    b1 = conv_b[DI + tid];
  }

  const size_t rowbase = (size_t)seq * NSEQ;
  for (int t = 0; t < NSEQ; ++t) {
    __syncthreads();                       // WAR guard on xsh / red
    const size_t tok = rowbase + t;
    // causal conv + SiLU for own xBC channel(s)
    {
      float acc = b0;
#pragma unroll
      for (int i = 0; i < 4; ++i) {
        const int tt = t - 3 + i;
        if (tt >= 0) acc += zx[(rowbase + tt) * LDZ + DI + tid] * w0[i];
      }
      xsh[tid] = acc / (1.f + expf(-acc));
    }
    if (tid < EXTRA) {
      float acc = b1;
#pragma unroll
      for (int i = 0; i < 4; ++i) {
        const int tt = t - 3 + i;
        if (tt >= 0) acc += zx[(rowbase + tt) * LDZ + 2 * DI + tid] * w1[i];
      }
      xsh[DI + tid] = acc / (1.f + expf(-acc));
    }
    if (tid < NH) {
      float draw = zx[tok * LDZ + (2 * DI + EXTRA) + tid] + dt_bias[tid];
      float dt = (draw > 20.f) ? draw : log1pf(expf(draw));
      dts[tid] = dt;
      dAs[tid] = expf(dt * (-expf(A_log[tid])));
    }
    __syncthreads();

    const float xv = xsh[tid];
    const float dt = dts[h], dA = dAs[h];
    const float dtx = dt * xv;
    float y = 0.f;
#pragma unroll
    for (int n = 0; n < NS; ++n) {
      st[n] = st[n] * dA + dtx * xsh[DI + n];        // B
      y += st[n] * xsh[DI + NS + n];                 // C
    }
    y += Dh * xv;

    // gated RMSNorm: val = y*silu(z); val *= rsqrt(mean(val^2)+1e-5)*norm_w
    const float z = zx[tok * LDZ + tid];
    float val = y * (z / (1.f + expf(-z)));
    float s = val * val;
#pragma unroll
    for (int off = 16; off; off >>= 1) s += __shfl_down(s, off, 32);
    if ((tid & 31) == 0) red[tid >> 5] = s;
    __syncthreads();
    float tot = 0.f;
#pragma unroll
    for (int i = 0; i < DI / 32; ++i) tot += red[i];
    const float inv = rsqrtf(tot / (float)DI + 1e-5f);
    yn[tok * DI + tid] = val * inv * nw;
  }
}

// ---------------------------------------------------------------------------
// Small helper kernels
// ---------------------------------------------------------------------------
__global__ void embed_kernel(const int* __restrict__ ids,
                             const float* __restrict__ emb,
                             float* __restrict__ x) {
  const int t = blockIdx.x, c = threadIdx.x;        // C = 128
  x[(size_t)t * 128 + c] = emb[(size_t)ids[t] * 128 + c];
}

__global__ void patch_mask_kernel(const float* __restrict__ xloc,
                                  const int* __restrict__ amask,
                                  const int* __restrict__ mstrat,
                                  const float* __restrict__ mask_emb,
                                  float* __restrict__ pm) {
  const int row = blockIdx.x, c = threadIdx.x;      // rows = B*N, C = 128
  const bool mp = amask[row] != 0;
  float v = xloc[((size_t)row * 8 + 7) * 128 + c];  // last byte of patch
  if (mp && mstrat[row] == 0) v = mask_emb[c];
  if (mp && mstrat[row] == 1) v = 0.f;              // never taken (strategy==0)
  pm[(size_t)row * 128 + c] = v;
}

__global__ void flip_seq_kernel(const float* __restrict__ in,
                                float* __restrict__ out, int Ln, int C) {
  const int row = blockIdx.x;
  const int b = row / Ln, l = row % Ln;
  const size_t src = (size_t)(b * Ln + (Ln - 1 - l)) * C;
  for (int c = threadIdx.x; c < C; c += blockDim.x)
    out[(size_t)row * C + c] = in[src + c];
}

__global__ void rmsnorm_offset_kernel(const float* __restrict__ x,
                                      const float* __restrict__ w,
                                      float* __restrict__ out, int D) {
  __shared__ float red[8];
  const int row = blockIdx.x, tid = threadIdx.x;    // 256 threads
  float s = 0.f;
  for (int c = tid; c < D; c += blockDim.x) {
    const float v = x[(size_t)row * D + c];
    s += v * v;
  }
#pragma unroll
  for (int off = 16; off; off >>= 1) s += __shfl_down(s, off, 32);
  if ((tid & 31) == 0) red[tid >> 5] = s;
  __syncthreads();
  float tot = 0.f;
#pragma unroll
  for (int i = 0; i < 8; ++i) tot += red[i];
  const float inv = rsqrtf(tot / (float)D + 1e-6f);
  for (int c = tid; c < D; c += blockDim.x)
    out[(size_t)row * D + c] = x[(size_t)row * D + c] * inv * (1.f + w[c]);
}

__global__ void silu_gate_kernel(const float* __restrict__ gx,
                                 float* __restrict__ xx, int D) {
  const int row = blockIdx.x;
  for (int c = threadIdx.x; c < D; c += blockDim.x) {
    const float g = gx[(size_t)row * 2 * D + c];
    const float v = gx[(size_t)row * 2 * D + D + c];
    xx[(size_t)row * D + c] = (g / (1.f + expf(-g))) * v;
  }
}

// ---------------------------------------------------------------------------
// Host-side orchestration
// ---------------------------------------------------------------------------
static void gemm(hipStream_t s, const float* A, const float* W,
                 const float* bias, const float* resid, const float* scale,
                 float* out, int M, int N, int K, int ldo, int col_off,
                 int flipL) {
  dim3 g((N + 63) / 64, (M + 63) / 64), b(128);
  gemm_bf16_wmma<<<g, b, 0, s>>>(A, W, bias, resid, scale, out, M, N, K, ldo,
                                 col_off, flipL);
}

extern "C" void kernel_launch(void* const* d_in, const int* in_sizes, int n_in,
                              void* d_out, int out_size, void* d_ws,
                              size_t ws_size, hipStream_t stream) {
  (void)in_sizes; (void)n_in; (void)out_size; (void)ws_size;
  // Model constants
  const int B = 4, NA = 256, P = 8;
  const int T1 = B * NA * P;        // 8192 local tokens
  const int T2 = B * NA;            // 1024 global tokens
  const int PD = 128, DM = 512;
  const int DI_L = 256, LDZ_L = 548;   // local:  2*256+32+4
  const int DI_G = 1024, LDZ_G = 2096; // global: 2*1024+32+16

  // ---- inputs (setup_inputs dict flatten order) ----
  const int*  byte_ids = (const int*)d_in[0];
  const int*  amask    = (const int*)d_in[1];
  const int*  mstrat   = (const int*)d_in[2];
  const float* byte_emb = (const float*)d_in[3];
  const float* mask_emb = (const float*)d_in[4];
  // local layers: base 5 + l*9 : {in_proj, conv_w, conv_b, dt_bias, A_log, D,
  //                               norm_w, out_proj, layer_scale}
  const float* patch_w = (const float*)d_in[41];
  const float* patch_b = (const float*)d_in[42];
  // global layers: base 43 + g*18 : fwd(8) bwd(8) out_proj layer_scale
  const float* final_nw = (const float*)d_in[151];
  const float* d1_w = (const float*)d_in[152];
  const float* d1_b = (const float*)d_in[153];
  const float* d2_w = (const float*)d_in[154];
  const float* d2_b = (const float*)d_in[155];
  const float* mlm_nw = (const float*)d_in[156];
  const float* dec_w = (const float*)d_in[157];
  const float* dec_b = (const float*)d_in[158];

  // ---- workspace carve ----
  float* ws = (float*)d_ws;
  size_t off = 0;
  float* xloc  = ws + off; off += (size_t)T1 * PD;     // 8192x128
  float* zxl   = ws + off; off += (size_t)T1 * LDZ_L;  // 8192x548
  float* ynl   = ws + off; off += (size_t)T1 * DI_L;   // 8192x256
  float* h     = ws + off; off += (size_t)T2 * DM;     // 1024x512
  float* hflip = ws + off; off += (size_t)T2 * DM;
  float* zxg   = ws + off; off += (size_t)T2 * LDZ_G;  // 1024x2096
  float* yng   = ws + off; off += (size_t)T2 * DI_G;   // 1024x1024
  float* cat   = ws + off; off += (size_t)T2 * 2 * DM; // 1024x1024
  // reuse zxl after local stage for MLM temporaries (3.1M <= 4.5M floats)
  float* gx  = zxl;
  float* xx  = gx  + (size_t)T2 * 2 * DM;
  float* xx2 = xx  + (size_t)T2 * DM;
  float* hn  = xx2 + (size_t)T2 * DM;
  float* xx3 = hn  + (size_t)T2 * DM;
  // reuse ynl for the masked patch buffer
  float* pm = ynl;

  // ---- 1. byte embedding ----
  embed_kernel<<<T1, PD, 0, stream>>>(byte_ids, byte_emb, xloc);

  // ---- 2. local Mamba2 layers ----
  for (int l = 0; l < 4; ++l) {
    const int pb = 5 + l * 9;
    const float* in_proj  = (const float*)d_in[pb + 0];
    const float* conv_w   = (const float*)d_in[pb + 1];
    const float* conv_b   = (const float*)d_in[pb + 2];
    const float* dt_bias  = (const float*)d_in[pb + 3];
    const float* A_log    = (const float*)d_in[pb + 4];
    const float* Dp       = (const float*)d_in[pb + 5];
    const float* norm_w   = (const float*)d_in[pb + 6];
    const float* out_proj = (const float*)d_in[pb + 7];
    const float* lscale   = (const float*)d_in[pb + 8];

    gemm(stream, xloc, in_proj, nullptr, nullptr, nullptr, zxl,
         T1, LDZ_L, PD, LDZ_L, 0, 0);
    scan_kernel<DI_L, P, LDZ_L, 4><<<B * NA, DI_L, 0, stream>>>(
        zxl, conv_w, conv_b, dt_bias, A_log, Dp, norm_w, ynl);
    gemm(stream, ynl, out_proj, nullptr, xloc, lscale, xloc,
         T1, PD, DI_L, PD, 0, 0);   // xloc += (yn @ W^T) * layer_scale
  }

  // ---- 3. patch extraction + masking + projection ----
  patch_mask_kernel<<<T2, PD, 0, stream>>>(xloc, amask, mstrat, mask_emb, pm);
  gemm(stream, pm, patch_w, patch_b, nullptr, nullptr, h,
       T2, DM, PD, DM, 0, 0);

  // ---- 4. global bidirectional Mamba2 layers ----
  for (int g = 0; g < 6; ++g) {
    const int gb = 43 + g * 18;
    const float* g_out   = (const float*)d_in[gb + 16];
    const float* g_scale = (const float*)d_in[gb + 17];
    for (int dir = 0; dir < 2; ++dir) {
      const int mb = gb + dir * 8;
      const float* in_proj  = (const float*)d_in[mb + 0];
      const float* conv_w   = (const float*)d_in[mb + 1];
      const float* conv_b   = (const float*)d_in[mb + 2];
      const float* dt_bias  = (const float*)d_in[mb + 3];
      const float* A_log    = (const float*)d_in[mb + 4];
      const float* Dp       = (const float*)d_in[mb + 5];
      const float* norm_w   = (const float*)d_in[mb + 6];
      const float* out_proj = (const float*)d_in[mb + 7];

      const float* src = h;
      if (dir == 1) {
        flip_seq_kernel<<<T2, 128, 0, stream>>>(h, hflip, NA, DM);
        src = hflip;
      }
      gemm(stream, src, in_proj, nullptr, nullptr, nullptr, zxg,
           T2, LDZ_G, DM, LDZ_G, 0, 0);
      scan_kernel<DI_G, 256, LDZ_G, 16><<<B, DI_G, 0, stream>>>(
          zxg, conv_w, conv_b, dt_bias, A_log, Dp, norm_w, yng);
      // write into concat buffer; bwd direction un-flips rows
      gemm(stream, yng, out_proj, nullptr, nullptr, nullptr, cat,
           T2, DM, DI_G, 2 * DM, dir == 0 ? 0 : DM, dir == 0 ? 0 : NA);
    }
    // h += (cat @ g_out^T) * layer_scale
    gemm(stream, cat, g_out, nullptr, h, g_scale, h,
         T2, DM, 2 * DM, DM, 0, 0);
  }

  // ---- 5. final norm + MLM head ----
  rmsnorm_offset_kernel<<<T2, 256, 0, stream>>>(h, final_nw, hn, DM);
  gemm(stream, hn, d1_w, d1_b, nullptr, nullptr, gx, T2, 2 * DM, DM,
       2 * DM, 0, 0);
  silu_gate_kernel<<<T2, 256, 0, stream>>>(gx, xx, DM);
  gemm(stream, xx, d2_w, d2_b, nullptr, nullptr, xx2, T2, DM, DM, DM, 0, 0);
  rmsnorm_offset_kernel<<<T2, 256, 0, stream>>>(xx2, mlm_nw, xx3, DM);
  gemm(stream, xx3, dec_w, dec_b, nullptr, nullptr, (float*)d_out,
       T2, 2056, DM, 2056, 0, 0);
}
